// OuterProductMeanSS_40793599378154
// MI455X (gfx1250) — matrix operand order repaired and checked by reading, not compiled, not run on hardware
//
#include <hip/hip_runtime.h>

// ---------------- problem constants ----------------
#define L_SEQ 512
#define C_S   384
#define C_H   32
#define C_Z   128
#define EPSF  1e-5f

// ---------------- WMMA / TDM types ----------------
typedef __attribute__((ext_vector_type(16))) __bf16 v16bf;
typedef __attribute__((ext_vector_type(8)))  float  v8f;
typedef __attribute__((ext_vector_type(4)))  unsigned int v4u;
typedef __attribute__((ext_vector_type(8)))  int    v8i;
typedef __attribute__((ext_vector_type(4)))  int    v4i;

union Frag32B {          // 32-byte WMMA operand fragment (8 VGPRs / lane)
    v16bf v;
    uint4 q[2];
    unsigned short u[16];
};

static __device__ __forceinline__ unsigned short f32_to_bf16(float f) {
    unsigned int x = __float_as_uint(f);
    x += 0x7FFFu + ((x >> 16) & 1u);   // round-to-nearest-even
    return (unsigned short)(x >> 16);
}

// ---------------------------------------------------------------------
// Tensor Data Mover: 1-D contiguous DMA global -> LDS.
// n8 = number of 8-byte elements (data_size=3). D# per CDNA5 ISA §8:
//  group0: [1:0]count=1 | [63:32]lds_addr | [120:64]global_addr | [127:126]type=2
//  group1: [17:16]data_size=3, [79:48]tensor_dim0=n8, [111:80]tensor_dim1=1,
//          [127:112]tile_dim0=n8 (1-D tile), [207:160]stride0=n8
//  groups 2/3: zero (tensor <= 2-D)
// Toolchain here exposes the 6-arg builtin:
//  (v4u g0, v8i g1, v4i g2, v4i g3, v8i extra, i32 cpol)
// ---------------------------------------------------------------------
static __device__ __forceinline__ void tdm_load_1d(unsigned lds_off,
                                                   const void* gptr,
                                                   unsigned n8) {
    unsigned long long ga = (unsigned long long)(uintptr_t)gptr;
    v4u g0;
    g0[0] = 1u;                                            // count=1, user mode
    g0[1] = lds_off;                                       // LDS byte address
    g0[2] = (unsigned)(ga & 0xFFFFFFFFu);                  // global_addr[31:0]
    g0[3] = (unsigned)((ga >> 32) & 0x01FFFFFFu) | 0x80000000u; // [56:32] | type=2
    v8i g1;
    g1[0] = (int)(3u << 16);                               // data_size = 8 bytes
    g1[1] = (int)((n8 & 0xFFFFu) << 16);                   // tensor_dim0[15:0]
    g1[2] = (int)(((n8 >> 16) & 0xFFFFu) | (1u << 16));    // td0[31:16] | td1[15:0]=1
    g1[3] = (int)((n8 & 0xFFFFu) << 16);                   // td1[31:16]=0 | tile_dim0
    g1[4] = 0;                                             // tile_dim1/2 unused
    g1[5] = (int)n8;                                       // tensor_dim0_stride lo
    g1[6] = 0;
    g1[7] = 0;
    v4i zz4 = {0, 0, 0, 0};
    v8i zz8 = {0, 0, 0, 0, 0, 0, 0, 0};
    __builtin_amdgcn_tensor_load_to_lds(g0, g1, zz4, zz4, zz8, 0);
}

// =====================================================================
// Kernel 1: LayerNorm over C_S + both 384->32 projections, emit bf16.
// One wave per sequence row; 8 waves / block.
// =====================================================================
__global__ __launch_bounds__(256) void k_lnproj(
    const float* __restrict__ s, const float* __restrict__ gamma,
    const float* __restrict__ beta,
    const float* __restrict__ w1, const float* __restrict__ b1,
    const float* __restrict__ w2, const float* __restrict__ b2,
    unsigned short* __restrict__ a_bf, unsigned short* __restrict__ b_bf)
{
    __shared__ float sn[8][C_S];
    const int wave = threadIdx.x >> 5;
    const int lane = threadIdx.x & 31;
    const int row  = blockIdx.x * 8 + wave;

    const float* srow = s + row * C_S;
    float sum = 0.f, sumsq = 0.f;
    for (int c = lane; c < C_S; c += 32) {
        float v = srow[c];
        sum += v; sumsq += v * v;
    }
    #pragma unroll
    for (int off = 16; off > 0; off >>= 1) {
        sum   += __shfl_xor(sum,   off, 32);
        sumsq += __shfl_xor(sumsq, off, 32);
    }
    const float mu   = sum * (1.f / C_S);
    const float var  = sumsq * (1.f / C_S) - mu * mu;
    const float rstd = rsqrtf(var + EPSF);

    for (int c = lane; c < C_S; c += 32)
        sn[wave][c] = (srow[c] - mu) * rstd * gamma[c] + beta[c];
    __syncthreads();

    // each lane owns one hidden channel h = lane
    float acc1 = b1[lane], acc2 = b2[lane];
    #pragma unroll 4
    for (int c = 0; c < C_S; ++c) {
        const float v = sn[wave][c];
        acc1 = fmaf(v, w1[c * C_H + lane], acc1);
        acc2 = fmaf(v, w2[c * C_H + lane], acc2);
    }
    a_bf[row * C_H + lane] = f32_to_bf16(acc1);
    b_bf[row * C_H + lane] = f32_to_bf16(acc2);
}

// =====================================================================
// Kernel 1b: convert W_out (1024x128 fp32) to bf16 once.
// =====================================================================
__global__ __launch_bounds__(256) void k_cvt(
    const float* __restrict__ w, unsigned short* __restrict__ wb, int n)
{
    int i = blockIdx.x * 256 + threadIdx.x;
    if (i < n) wb[i] = f32_to_bf16(w[i]);
}

// =====================================================================
// Kernel 2: m[j,a,z] = sum_b b[j,b] * W[a*32+b, z]
// GEMM tiles: 32 j-tiles x 32 a x 8 z-tiles = 8192 WMMAs.
// block = (j_tile, a); wave = z-tile.
// =====================================================================
__global__ __launch_bounds__(256) void k_stage2(
    const unsigned short* __restrict__ b_bf,
    const unsigned short* __restrict__ w_bf,
    unsigned short* __restrict__ m_bf)
{
    const int wave = threadIdx.x >> 5;
    const int lane = threadIdx.x & 31;
    const int jt   = blockIdx.x >> 5;   // 0..31
    const int a    = blockIdx.x & 31;   // 0..31
    const int z0   = wave * 16;
    const int half = lane >> 4;
    const int mrow = lane & 15;
    const int zc   = z0 + mrow;

    // A fragment: rows j = jt*16+mrow, K = b (0..31)
    Frag32B A;
    const unsigned short* arow = b_bf + (jt * 16 + mrow) * C_H;
    A.q[0] = *(const uint4*)(arow + half * 16);
    A.q[1] = *(const uint4*)(arow + half * 16 + 8);

    // B fragment: W[b][z] = w_bf[(a*32 + b)*128 + z]
    Frag32B B;
    #pragma unroll
    for (int v = 0; v < 8; ++v) {
        const int k = half * 16 + 2 * v;
        B.u[2 * v]     = w_bf[(a * C_H + k)     * C_Z + zc];
        B.u[2 * v + 1] = w_bf[(a * C_H + k + 1) * C_Z + zc];
    }

    v8f c = {};
    c = __builtin_amdgcn_wmma_f32_16x16x32_bf16(false, A.v, false, B.v,
                                                (short)0, c, false, false);
    #pragma unroll
    for (int r = 0; r < 8; ++r) {
        const int j = jt * 16 + r + 8 * half;
        m_bf[(j * C_H + a) * C_Z + zc] = f32_to_bf16(c[r]);
    }
}

// =====================================================================
// Kernel 3: z[i,j,z] = sum_a a[i,a] * m[j,a,z] + b_out[z]
// One block per j. TDM stages all of A (32KB, LDS off 0) and M_j
// (8KB, LDS off 32768); wave0/wave1 issue the DMAs concurrently,
// s_wait_tensorcnt, then barrier. 8 waves: wave owns z-tile, loops
// 32 i-tiles -> 256 WMMAs/block.
// =====================================================================
__global__ __launch_bounds__(256) void k_stage3(
    const unsigned short* __restrict__ a_bf,
    const unsigned short* __restrict__ m_bf,
    const float* __restrict__ bias_z,
    float* __restrict__ out)
{
    // single shared buffer so TDM lds_addr offsets (0 / 32768) match layout
    __shared__ __align__(16) unsigned short lbuf[L_SEQ * C_H + C_H * C_Z]; // 40 KB
    unsigned short* lA = lbuf;                 // LDS byte offset 0
    unsigned short* lM = lbuf + L_SEQ * C_H;   // LDS byte offset 32768

    const int j   = blockIdx.x;
    const int tid = threadIdx.x;
    const int wv  = tid >> 5;

    if (wv == 0) {                                  // DMA A: 32 KB = 4096 x 8B
        tdm_load_1d(0u, a_bf, (L_SEQ * C_H * 2) / 8);
        __builtin_amdgcn_s_wait_tensorcnt(0);
    } else if (wv == 1) {                           // DMA M_j: 8 KB = 1024 x 8B
        tdm_load_1d((unsigned)(L_SEQ * C_H * 2), m_bf + (size_t)j * C_H * C_Z,
                    (C_H * C_Z * 2) / 8);
        __builtin_amdgcn_s_wait_tensorcnt(0);
    }
    __syncthreads();

    const int lane = tid & 31;
    const int half = lane >> 4;
    const int mrow = lane & 15;
    const int z0   = wv * 16;
    const int zc   = z0 + mrow;

    // B fragment (M_j[a][z]) built once from LDS, reused for 32 WMMAs
    Frag32B B;
    #pragma unroll
    for (int v = 0; v < 8; ++v) {
        const int k = half * 16 + 2 * v;
        B.u[2 * v]     = lM[k * C_Z + zc];
        B.u[2 * v + 1] = lM[(k + 1) * C_Z + zc];
    }
    const float bias = bias_z[zc];

    for (int it = 0; it < L_SEQ / 16; ++it) {
        Frag32B A;
        const unsigned short* arow = lA + (it * 16 + mrow) * C_H;
        A.q[0] = *(const uint4*)(arow + half * 16);        // ds_load_b128
        A.q[1] = *(const uint4*)(arow + half * 16 + 8);   // ds_load_b128
        v8f c = {};
        c = __builtin_amdgcn_wmma_f32_16x16x32_bf16(false, A.v, false, B.v,
                                                    (short)0, c, false, false);
        #pragma unroll
        for (int r = 0; r < 8; ++r) {
            const int i = it * 16 + r + 8 * half;
            out[(size_t)i * (L_SEQ * C_Z) + (size_t)j * C_Z + zc] = c[r] + bias;
        }
    }
}

// =====================================================================
// Host launcher
// =====================================================================
extern "C" void kernel_launch(void* const* d_in, const int* in_sizes, int n_in,
                              void* d_out, int out_size, void* d_ws, size_t ws_size,
                              hipStream_t stream) {
    const float* s     = (const float*)d_in[0];
    const float* gamma = (const float*)d_in[1];
    const float* beta  = (const float*)d_in[2];
    const float* w1    = (const float*)d_in[3];
    const float* b1    = (const float*)d_in[4];
    const float* w2    = (const float*)d_in[5];
    const float* b2    = (const float*)d_in[6];
    const float* w_out = (const float*)d_in[7];
    const float* b_out = (const float*)d_in[8];
    float* out = (float*)d_out;

    // workspace partition (all bf16 stored as u16); total ~4.5 MB
    unsigned short* a_bf = (unsigned short*)d_ws;          // 512*32
    unsigned short* b_bf = a_bf + L_SEQ * C_H;             // 512*32
    unsigned short* w_bf = b_bf + L_SEQ * C_H;             // 1024*128
    unsigned short* m_bf = w_bf + C_H * C_H * C_Z;         // 512*32*128

    k_lnproj<<<L_SEQ / 8, 256, 0, stream>>>(s, gamma, beta, w1, b1, w2, b2,
                                            a_bf, b_bf);
    k_cvt<<<(C_H * C_H * C_Z + 255) / 256, 256, 0, stream>>>(w_out, w_bf,
                                                             C_H * C_H * C_Z);
    k_stage2<<<(L_SEQ / 16) * C_H, 256, 0, stream>>>(b_bf, w_bf, m_bf);
    k_stage3<<<L_SEQ, 256, 0, stream>>>(a_bf, m_bf, b_out, out);
}
